// MultiHeadAttention_7739531067803
// MI455X (gfx1250) — matrix operand order repaired and checked by reading, not compiled, not run on hardware
//
#include <hip/hip_runtime.h>
#include <hip/hip_bf16.h>
#include <stdint.h>

// MI455X / gfx1250: wave32, WMMA bf16 16x16x32 (f32 accum), async global->LDS.
// Workspace layout (bf16 elements), SZ = H*B*N*64 = 8388608:
//   [0]Q_node [1]Q_pos [2]K_node [3]K_pos
//   [4]V_node0 [5]V_node1 [6]V_pos0 [7]V_pos1
//   [8..9] heads_node [H][B][N][128]   [10..11] heads_pos
// Total = 12*SZ*2 bytes = 201,326,592 bytes of d_ws.

typedef __bf16 bf16_t;
typedef __attribute__((ext_vector_type(16))) __bf16 v16bf;
typedef __attribute__((ext_vector_type(8)))  float  v8f;

#define B_   16
#define N_   1024
#define D_   512
#define H_   8
#define KD_  64
#define E_   512
#define NORM_ 0.125f

static constexpr size_t SZ = (size_t)H_ * B_ * N_ * KD_;

static __device__ __forceinline__ bf16_t f2bf(float f) {
  unsigned u = __builtin_bit_cast(unsigned, f);
  unsigned r = u + 0x7FFFu + ((u >> 16) & 1u);       // round-to-nearest-even
  unsigned short hs = (unsigned short)(r >> 16);
  return __builtin_bit_cast(bf16_t, hs);
}

// 16 contiguous bf16 from LDS -> B-fragment register.
static __device__ __forceinline__ v16bf ld_b16(const bf16_t* p) {
  v16bf v;
#pragma unroll
  for (int j = 0; j < 16; ++j) v[j] = p[j];
  return v;
}

// Async DMA: 16 bytes global -> LDS, tracked by ASYNCcnt (no VGPR data path).
// Both addresses must be 16-byte aligned.
static __device__ __forceinline__ void async_b128(bf16_t* lds_dst, const bf16_t* gsrc) {
  unsigned l = (unsigned)(size_t)(void*)lds_dst;            // LDS byte offset
  unsigned long long g = (unsigned long long)(size_t)(const void*)gsrc;
  asm volatile("global_load_async_to_lds_b128 %0, %1, off"
               :: "v"(l), "v"(g) : "memory");
}
static __device__ __forceinline__ void wait_async0() {
  asm volatile("s_wait_asynccnt 0x0" ::: "memory");
}

// ---------------------------------------------------------------------------
// Kernel 1: all 64 head-projections. grid (mtile=8, b=16, g=64), 256 threads.
// C[1024x64] = A[1024x512] x W[512x64], f32 inputs -> bf16 operands -> bf16 out.
// (f32->bf16 conversion happens while staging, so this one keeps VALU staging.)
// ---------------------------------------------------------------------------
__global__ __launch_bounds__(256) void proj_kernel(
    const float* __restrict__ hn, const float* __restrict__ hp,
    const float* __restrict__ wqn, const float* __restrict__ wqp,
    const float* __restrict__ wkn, const float* __restrict__ wkp,
    const float* __restrict__ wvn, const float* __restrict__ wvp,
    bf16_t* __restrict__ ws)
{
  __shared__ bf16_t lA[128 * 40];   // A tile 128x32 (stride 40)
  __shared__ bf16_t lBt[64 * 40];   // W tile transposed: [col 64][k 32]

  const int mt = blockIdx.x, b = blockIdx.y, g = blockIdx.z;
  const float* A; const float* W; bf16_t* Out;
  if (g < 32) {                     // Q_node / Q_pos / K_node / K_pos
    int kind = g >> 3, h = g & 7;
    A = (kind & 1) ? hp : hn;
    W = (kind == 0 ? wqn : kind == 1 ? wqp : kind == 2 ? wkn : wkp)
        + (size_t)h * D_ * KD_;
    Out = ws + (size_t)kind * SZ + ((size_t)(h * B_ + b) * N_) * KD_;
  } else {                          // V_node (2H) / V_pos (2H), reshape (2,H)
    int hv = (g - 32) & 15, which = (g - 32) >> 4;
    int s = hv >> 3, h = hv & 7;
    A = which ? hp : hn;
    W = (which ? wvp : wvn) + (size_t)hv * D_ * KD_;
    Out = ws + (size_t)(4 + 2 * which + s) * SZ
             + ((size_t)(h * B_ + b) * N_) * KD_;
  }
  A += (size_t)b * N_ * D_ + (size_t)(mt * 128) * D_;

  const int tid = threadIdx.x;
  const int lane = tid & 31, wave = tid >> 5;
  const int lrow = lane & 15, lhalf = lane >> 4;
  const int row16 = wave * 16;

  v8f c[4] = {};

  for (int k0 = 0; k0 < D_; k0 += 32) {
    for (int i = tid; i < 1024; i += 256) {            // A tile 128x32 f32
      int r = i >> 3, c4 = (i & 7) * 4;
      float4 v = *(const float4*)(A + (size_t)r * D_ + k0 + c4);
      bf16_t* d = &lA[r * 40 + c4];
      d[0] = f2bf(v.x); d[1] = f2bf(v.y); d[2] = f2bf(v.z); d[3] = f2bf(v.w);
    }
    for (int i = tid; i < 512; i += 256) {             // W tile 32x64 -> T
      int r = i >> 4, c4 = (i & 15) * 4;
      float4 v = *(const float4*)(W + (size_t)(k0 + r) * KD_ + c4);
      lBt[(c4 + 0) * 40 + r] = f2bf(v.x);
      lBt[(c4 + 1) * 40 + r] = f2bf(v.y);
      lBt[(c4 + 2) * 40 + r] = f2bf(v.z);
      lBt[(c4 + 3) * 40 + r] = f2bf(v.w);
    }
    __syncthreads();

    // preload all fragments, then issue WMMAs back-to-back
    v16bf a;
    {
      const bf16_t* Ap = &lA[(row16 + lrow) * 40];
      int kb = lhalf * 16;
#pragma unroll
      for (int j = 0; j < 8; ++j) { a[j] = Ap[kb + j]; a[8 + j] = Ap[8 + kb + j]; }
    }
    v16bf bb[4];
#pragma unroll
    for (int t = 0; t < 4; ++t)
      bb[t] = ld_b16(&lBt[(t * 16 + lrow) * 40 + lhalf * 16]);
#pragma unroll
    for (int t = 0; t < 4; ++t)
      c[t] = __builtin_amdgcn_wmma_f32_16x16x32_bf16(
                 false, a, false, bb[t], (short)0, c[t], false, false);
    __syncthreads();
  }

#pragma unroll
  for (int t = 0; t < 4; ++t)
#pragma unroll
    for (int r = 0; r < 8; ++r)
      Out[(size_t)(mt * 128 + row16 + r + lhalf * 8) * KD_ + t * 16 + lrow]
          = f2bf(c[t][r]);
}

// ---------------------------------------------------------------------------
// Kernel 2: flash attention. grid (qtile=8, b=16, h*2+s=16), 256 threads.
// K chunk staged via GLOBAL_LOAD_ASYNC_TO_LDS_B128 (pure bf16 copy);
// V chunks transposed on the fly (VALU path).
// ---------------------------------------------------------------------------
__global__ __launch_bounds__(256) void attn_kernel(bf16_t* __restrict__ ws)
{
  __shared__ bf16_t lK[32 * 72];        // K chunk [key 32][feat 64] (16B-aligned rows)
  __shared__ bf16_t lVn[64 * 40];       // V_node chunk transposed [feat][key]
  __shared__ bf16_t lVp[64 * 40];       // V_pos  chunk transposed [feat][key]
  __shared__ bf16_t lP[8 * 16 * 40];    // per-wave P tiles 16x32

  const int qt = blockIdx.x, b = blockIdx.y;
  const int h = blockIdx.z >> 1, s = blockIdx.z & 1;
  const size_t hb = (size_t)(h * B_ + b) * N_;
  const bf16_t* Q  = ws + (size_t)(0 + s) * SZ + hb * KD_;
  const bf16_t* K  = ws + (size_t)(2 + s) * SZ + hb * KD_;
  const bf16_t* Vn = ws + (size_t)(4 + s) * SZ + hb * KD_;
  const bf16_t* Vp = ws + (size_t)(6 + s) * SZ + hb * KD_;
  bf16_t* On = ws + 8 * SZ  + hb * 128 + s * 64;
  bf16_t* Op = ws + 10 * SZ + hb * 128 + s * 64;

  const int tid = threadIdx.x, lane = tid & 31, wave = tid >> 5;
  const int lrow = lane & 15, lhalf = lane >> 4;
  const int row16 = wave * 16, q0 = qt * 128;

  // Q A-fragments for this wave's 16 query rows (contraction 64 = 2 frags)
  v16bf aq0, aq1;
  {
    const bf16_t* Qrow = Q + (size_t)(q0 + row16 + lrow) * KD_;
    int kb = lhalf * 16;
#pragma unroll
    for (int j = 0; j < 8; ++j) {
      aq0[j]     = Qrow[kb + j];      aq0[8 + j] = Qrow[8 + kb + j];
      aq1[j]     = Qrow[32 + kb + j]; aq1[8 + j] = Qrow[40 + kb + j];
    }
  }

  v8f on[4] = {}, op[4] = {};
  float m[8], l[8];
#pragma unroll
  for (int r = 0; r < 8; ++r) { m[r] = -3.0e38f; l[r] = 0.f; }

  bf16_t* Pw = &lP[wave * (16 * 40)];

  for (int j0 = 0; j0 < N_; j0 += 32) {
    {   // K chunk: 32 rows x 64 bf16 = 256 x 16B async DMAs, one per thread
      int r = tid >> 3, c8 = (tid & 7) * 8;
      async_b128(&lK[r * 72 + c8], K + (size_t)(j0 + r) * KD_ + c8);
    }
    for (int i = tid; i < 512; i += 256) {             // V chunks, transposed
      int kk = i >> 4, f4 = (i & 15) * 4;
      const bf16_t* vs = Vn + (size_t)(j0 + kk) * KD_ + f4;
#pragma unroll
      for (int j = 0; j < 4; ++j) lVn[(f4 + j) * 40 + kk] = vs[j];
      vs = Vp + (size_t)(j0 + kk) * KD_ + f4;
#pragma unroll
      for (int j = 0; j < 4; ++j) lVp[(f4 + j) * 40 + kk] = vs[j];
    }
    wait_async0();
    __syncthreads();

    // scores S = Q K^T : 16 x 32 per wave; preload fragments, then WMMA chain
    v16bf bk[4];
    bk[0] = ld_b16(&lK[lrow * 72 + lhalf * 16]);
    bk[1] = ld_b16(&lK[lrow * 72 + 32 + lhalf * 16]);
    bk[2] = ld_b16(&lK[(16 + lrow) * 72 + lhalf * 16]);
    bk[3] = ld_b16(&lK[(16 + lrow) * 72 + 32 + lhalf * 16]);
    v8f c0 = {}, c1 = {};
    c0 = __builtin_amdgcn_wmma_f32_16x16x32_bf16(false, aq0, false, bk[0], (short)0, c0, false, false);
    c0 = __builtin_amdgcn_wmma_f32_16x16x32_bf16(false, aq1, false, bk[1], (short)0, c0, false, false);
    c1 = __builtin_amdgcn_wmma_f32_16x16x32_bf16(false, aq0, false, bk[2], (short)0, c1, false, false);
    c1 = __builtin_amdgcn_wmma_f32_16x16x32_bf16(false, aq1, false, bk[3], (short)0, c1, false, false);

    // online softmax: rows map to M = r + 8*lhalf, cols to lane&15 (+16 for c1)
    float alpha[8];
#pragma unroll
    for (int r = 0; r < 8; ++r) {
      float s0 = c0[r] * NORM_, s1 = c1[r] * NORM_;
      float rm = fmaxf(s0, s1);
#pragma unroll
      for (int off = 1; off < 16; off <<= 1)
        rm = fmaxf(rm, __shfl_xor(rm, off, 32));
      float mn = fmaxf(m[r], rm);
      alpha[r] = __expf(m[r] - mn);
      m[r] = mn;
      float p0 = __expf(s0 - mn), p1 = __expf(s1 - mn);
      float rs = p0 + p1;
#pragma unroll
      for (int off = 1; off < 16; off <<= 1)
        rs += __shfl_xor(rs, off, 32);
      l[r] = l[r] * alpha[r] + rs;
      Pw[(r + lhalf * 8) * 40 + lrow]      = f2bf(p0);   // C-layout -> LDS
      Pw[(r + lhalf * 8) * 40 + 16 + lrow] = f2bf(p1);
    }
#pragma unroll
    for (int t = 0; t < 4; ++t)
#pragma unroll
      for (int r = 0; r < 8; ++r) { on[t][r] *= alpha[r]; op[t][r] *= alpha[r]; }

    // reload P as an A-fragment (wave-private LDS tile, no barrier needed)
    v16bf ap;
    {
      const bf16_t* Pp = &Pw[lrow * 40];
      int kb = lhalf * 16;
#pragma unroll
      for (int j = 0; j < 8; ++j) { ap[j] = Pp[kb + j]; ap[8 + j] = Pp[8 + kb + j]; }
    }
    // preload all 8 V fragments, then issue the 8 WMMAs back-to-back
    v16bf bv[8];
#pragma unroll
    for (int t = 0; t < 4; ++t) {
      bv[t]     = ld_b16(&lVn[(t * 16 + lrow) * 40 + lhalf * 16]);
      bv[4 + t] = ld_b16(&lVp[(t * 16 + lrow) * 40 + lhalf * 16]);
    }
#pragma unroll
    for (int t = 0; t < 4; ++t)
      on[t] = __builtin_amdgcn_wmma_f32_16x16x32_bf16(false, ap, false, bv[t], (short)0, on[t], false, false);
#pragma unroll
    for (int t = 0; t < 4; ++t)
      op[t] = __builtin_amdgcn_wmma_f32_16x16x32_bf16(false, ap, false, bv[4 + t], (short)0, op[t], false, false);
    __syncthreads();
  }

  float inv[8];
#pragma unroll
  for (int r = 0; r < 8; ++r) inv[r] = 1.0f / l[r];
#pragma unroll
  for (int t = 0; t < 4; ++t)
#pragma unroll
    for (int r = 0; r < 8; ++r) {
      size_t ro = (size_t)(q0 + row16 + r + lhalf * 8) * 128 + t * 16 + lrow;
      On[ro] = f2bf(on[t][r] * inv[r]);
      Op[ro] = f2bf(op[t][r] * inv[r]);
    }
}

// ---------------------------------------------------------------------------
// Kernel 3: head-summed output GEMM. grid (mtile=8, etile=4, which*16+b = 32).
// out[b] (1024x512) = sum_h heads[h,b] (1024x128) x W_out[h] (128x512)
// heads tile staged via async DMA; W tile converted f32->bf16 via VALU.
// ---------------------------------------------------------------------------
__global__ __launch_bounds__(256) void out_kernel(
    const float* __restrict__ won, const float* __restrict__ wop,
    const bf16_t* __restrict__ ws, float* __restrict__ out)
{
  __shared__ bf16_t lA[128 * 48];    // heads tile 128x32 (stride 48: 96B rows)
  __shared__ bf16_t lBt[128 * 40];   // W tile transposed [col 128][k 32]

  const int mt = blockIdx.x, et = blockIdx.y;
  const int which = blockIdx.z >> 4, b = blockIdx.z & 15;
  const bf16_t* Hd = ws + (size_t)(8 + 2 * which) * SZ;
  const float* W = which ? wop : won;
  float* O = out + (size_t)which * B_ * N_ * E_ + (size_t)b * N_ * E_;

  const int tid = threadIdx.x, lane = tid & 31, wave = tid >> 5;
  const int lrow = lane & 15, lhalf = lane >> 4;
  const int row16 = wave * 16;

  v8f c[8] = {};

  for (int h = 0; h < H_; ++h) {
    const bf16_t* Ahb = Hd + ((size_t)(h * B_ + b) * N_ + mt * 128) * 128;
    const float* Wh = W + (size_t)h * 128 * E_ + et * 128;
    for (int kc = 0; kc < 128; kc += 32) {
      for (int i = tid; i < 512; i += 256) {           // heads tile: async DMA
        int r = i >> 2, c8 = (i & 3) * 8;
        async_b128(&lA[r * 48 + c8], Ahb + (size_t)r * 128 + kc + c8);
      }
      for (int i = tid; i < 1024; i += 256) {          // W tile f32 -> bf16, T
        int kk = i >> 5, f4 = (i & 31) * 4;
        float4 v = *(const float4*)(Wh + (size_t)(kc + kk) * E_ + f4);
        lBt[(f4 + 0) * 40 + kk] = f2bf(v.x);
        lBt[(f4 + 1) * 40 + kk] = f2bf(v.y);
        lBt[(f4 + 2) * 40 + kk] = f2bf(v.z);
        lBt[(f4 + 3) * 40 + kk] = f2bf(v.w);
      }
      wait_async0();
      __syncthreads();

      v16bf a;
      {
        const bf16_t* Ap = &lA[(row16 + lrow) * 48];
        int kb = lhalf * 16;
#pragma unroll
        for (int j = 0; j < 8; ++j) { a[j] = Ap[kb + j]; a[8 + j] = Ap[8 + kb + j]; }
      }
      v16bf bw[8];
#pragma unroll
      for (int t = 0; t < 8; ++t)
        bw[t] = ld_b16(&lBt[(t * 16 + lrow) * 40 + lhalf * 16]);
#pragma unroll
      for (int t = 0; t < 8; ++t)
        c[t] = __builtin_amdgcn_wmma_f32_16x16x32_bf16(
                   false, a, false, bw[t], (short)0, c[t], false, false);
      __syncthreads();
    }
  }
#pragma unroll
  for (int t = 0; t < 8; ++t)
#pragma unroll
    for (int r = 0; r < 8; ++r)
      O[(size_t)(mt * 128 + row16 + r + lhalf * 8) * E_ + et * 128 + t * 16 + lrow]
          = c[t][r];
}

// ---------------------------------------------------------------------------
extern "C" void kernel_launch(void* const* d_in, const int* in_sizes, int n_in,
                              void* d_out, int out_size, void* d_ws, size_t ws_size,
                              hipStream_t stream) {
  const float* hn  = (const float*)d_in[0];
  const float* hp  = (const float*)d_in[1];
  const float* wqn = (const float*)d_in[2];
  const float* wqp = (const float*)d_in[3];
  const float* wkn = (const float*)d_in[4];
  const float* wkp = (const float*)d_in[5];
  const float* wvn = (const float*)d_in[6];
  const float* wvp = (const float*)d_in[7];
  const float* won = (const float*)d_in[8];
  const float* wop = (const float*)d_in[9];
  bf16_t* ws = (bf16_t*)d_ws;
  float* out = (float*)d_out;

  proj_kernel<<<dim3(8, 16, 64), 256, 0, stream>>>(hn, hp, wqn, wqp,
                                                   wkn, wkp, wvn, wvp, ws);
  attn_kernel<<<dim3(8, 16, 16), 256, 0, stream>>>(ws);
  out_kernel<<<dim3(8, 4, 32), 256, 0, stream>>>(won, wop, ws, out);
}
